// TransformerLayer_62268435857813
// MI455X (gfx1250) — matrix operand order repaired and checked by reading, not compiled, not run on hardware
//
#include <hip/hip_runtime.h>
#include <hip/hip_bf16.h>

// ---------------------------------------------------------------------------
// TransformerLayer for MI455X (gfx1250): LN1 -> QKV -> GQA attn -> proj(+res)
// -> LN2 -> router/top2 -> dense MoE (gelu) -> combine.
// Big GEMMs: v_wmma_f32_16x16x32_bf16, 128x128 macro-tiles, LDS double
// buffering fed by global_load_async_to_lds_b128 (ASYNCcnt), fp32 accum.
// All LDS accesses use explicit address_space(3) pointers so fragment reads
// lower to ds_load_b128 (32-bit addressing), not flat.
// ---------------------------------------------------------------------------

typedef __bf16 bf16;
typedef __attribute__((ext_vector_type(16))) __bf16 v16bf;
typedef __attribute__((ext_vector_type(8)))  float  v8f;

typedef __attribute__((address_space(3))) __bf16 lds_bf16;
typedef __attribute__((address_space(3))) float  lds_float;

#define S_LEN  1024
#define BATCH  2
#define HDIM   1024
#define NHEADS 16
#define NKVH   4
#define HEADD  64
#define NEXP   8
#define FDIM   2048
#define TTOK   (S_LEN * BATCH)               /* 2048 tokens */
#define QKV_O  ((NHEADS + 2 * NKVH) * HEADD) /* 1536 */

#define LDSB 72                         /* LDS row stride (elems): 144B rows */
#define TILE_BYTES (128 * LDSB * 2)     /* one 128x64 bf16 tile in LDS */
#define GEMM_LDS (4 * TILE_BYTES)       /* A+B, double buffered = 73728 B */

// Low 32 bits of the flat shared-aperture address == DS offset (ISA 10.2).
__device__ __forceinline__ uint32_t lds_base(const void* p) {
  return (uint32_t)(uintptr_t)p;
}

// ---------------- WMMA fragment helpers (ISA 7.12.2 layouts) ---------------

__device__ __forceinline__ v16bf frag_a(uint32_t byteoff, int lda) {
  const int lane = threadIdx.x & 31;
  const int m    = lane & 15;
  const int koff = (lane & 16) ? 8 : 0;   // lanes 16-31 hold K {8-15,24-31}
  const lds_bf16* p =
      (const lds_bf16*)(uintptr_t)byteoff + m * lda + koff;
  v16bf f;
#pragma unroll
  for (int e = 0; e < 8; ++e) f[e] = p[e];
#pragma unroll
  for (int e = 0; e < 8; ++e) f[8 + e] = p[16 + e];
  return f;
}

// B staged in LDS as Bl[n][k] (K contiguous per B-column row).
__device__ __forceinline__ v16bf frag_b(uint32_t byteoff, int ldb) {
  const int lane = threadIdx.x & 31;
  const int n    = lane & 15;
  const int koff = (lane & 16) ? 8 : 0;   // lanes 16-31 hold K {8-15,24-31}
  const lds_bf16* p =
      (const lds_bf16*)(uintptr_t)byteoff + n * ldb + koff;
  v16bf f;
#pragma unroll
  for (int e = 0; e < 8; ++e) f[e] = p[e];
#pragma unroll
  for (int e = 0; e < 8; ++e) f[8 + e] = p[16 + e];
  return f;
}

__device__ __forceinline__ v8f wmma_bf16(v16bf a, v16bf b, v8f c) {
  return __builtin_amdgcn_wmma_f32_16x16x32_bf16(
      false, a, false, b, (short)0, c, false, false);
}

__device__ __forceinline__ void wait_async_lds() {
  asm volatile("s_wait_asynccnt 0" ::: "memory");
}

// Stage a 128x64 bf16 tile (rows K-contiguous in global) into LDS via the
// CDNA5 async-to-LDS path. 512 threads: thread -> row tid/4, 32B chunk.
// NOTE: IOFFSET applies to BOTH lds and global addresses (ISA async ops).
__device__ __forceinline__ void stage_async_128x64(const bf16* g, int ld,
                                                   uint32_t lds) {
  const int tid = threadIdx.x;
  const int r = tid >> 2;            // 0..127
  const int k = (tid & 3) << 4;      // 0,16,32,48
  const bf16* gp = g + (size_t)r * ld + k;
  const uint32_t la = lds + (uint32_t)(r * (LDSB * 2) + k * 2);
  asm volatile("global_load_async_to_lds_b128 %0, %1, off"
               :: "v"(la), "v"(gp) : "memory");
  asm volatile("global_load_async_to_lds_b128 %0, %1, off offset:16"
               :: "v"(la), "v"(gp) : "memory");
}

// ---------------- 128x128 GEMM core ----------------------------------------
// C[M,N] = A[M,K] x B[N,K]^T, both bf16 row-major with K contiguous.
// 512 threads / 16 waves; wave (wm,wn) owns a 32x32 block = 2x2 accumulators.
__device__ __forceinline__ void gemm_core_128(
    const bf16* __restrict__ A, int lda,
    const bf16* __restrict__ B, int ldb, int K,
    int rowBase, int colBase, uint32_t sbase, v8f acc[2][2]) {
  const int tid = threadIdx.x;
  const int w = tid >> 5, wm = w >> 2, wn = w & 3;
  const uint32_t Aoff[2] = {sbase, sbase + 2 * TILE_BYTES};
  const uint32_t Boff[2] = {sbase + TILE_BYTES, sbase + 3 * TILE_BYTES};
  const bf16* gA = A + (size_t)rowBase * lda;
  const bf16* gB = B + (size_t)colBase * ldb;

  stage_async_128x64(gA, lda, Aoff[0]);
  stage_async_128x64(gB, ldb, Boff[0]);
  wait_async_lds();
  __syncthreads();

  int cur = 0;
  const int nk = K >> 6;
  for (int kt = 0; kt < nk; ++kt) {
    if (kt + 1 < nk) {  // overlap next tile's async loads with the math
      stage_async_128x64(gA + (kt + 1) * 64, lda, Aoff[cur ^ 1]);
      stage_async_128x64(gB + (kt + 1) * 64, ldb, Boff[cur ^ 1]);
    }
    const uint32_t Ap = Aoff[cur] + (uint32_t)(wm * 32 * LDSB * 2);
    const uint32_t Bp = Boff[cur] + (uint32_t)(wn * 32 * LDSB * 2);
#pragma unroll
    for (int sk = 0; sk < 2; ++sk) {
      v16bf a0 = frag_a(Ap + sk * 64, LDSB);
      v16bf a1 = frag_a(Ap + 16 * LDSB * 2 + sk * 64, LDSB);
      v16bf b0 = frag_b(Bp + sk * 64, LDSB);
      v16bf b1 = frag_b(Bp + 16 * LDSB * 2 + sk * 64, LDSB);
      acc[0][0] = wmma_bf16(a0, b0, acc[0][0]);
      acc[0][1] = wmma_bf16(a0, b1, acc[0][1]);
      acc[1][0] = wmma_bf16(a1, b0, acc[1][0]);
      acc[1][1] = wmma_bf16(a1, b1, acc[1][1]);
    }
    wait_async_lds();
    __syncthreads();
    cur ^= 1;
  }
}

// D layout: element i -> row = i + (lane&16 ? 8:0), col = lane&15.
#define EPILOGUE_IDX                                        \
  const int lane = threadIdx.x & 31;                        \
  const int w = threadIdx.x >> 5, wm = w >> 2, wn = w & 3;  \
  const int nl = lane & 15;                                 \
  const int ml = (lane & 16) ? 8 : 0;

// ---------------- Elementwise / small kernels ------------------------------

__global__ void __launch_bounds__(256) cast_f32_bf16_kernel(
    const float* __restrict__ in, bf16* __restrict__ out, int n) {
  int i = blockIdx.x * 256 + threadIdx.x;
  if (i < n) out[i] = (bf16)in[i];
}

// Transpose+cast one [R,C] fp32 matrix (blockIdx.z selects expert) -> [C,R] bf16
__global__ void __launch_bounds__(256) transpose_cast_kernel(
    const float* __restrict__ in, bf16* __restrict__ out, int R, int C) {
  const float* src = in + (size_t)blockIdx.z * R * C;
  bf16* dst = out + (size_t)blockIdx.z * R * C;
  __shared__ float tile[32][33];
  const int c0 = blockIdx.x * 32, r0 = blockIdx.y * 32;
  const int tx = threadIdx.x & 31, ty = threadIdx.x >> 5;
#pragma unroll
  for (int i = 0; i < 32; i += 8)
    tile[ty + i][tx] = src[(size_t)(r0 + ty + i) * C + c0 + tx];
  __syncthreads();
#pragma unroll
  for (int i = 0; i < 32; i += 8)
    dst[(size_t)(c0 + ty + i) * R + r0 + tx] = (bf16)tile[tx][ty + i];
}

__global__ void __launch_bounds__(256) layernorm_kernel(
    const float* __restrict__ x, const float* __restrict__ w,
    const float* __restrict__ b, bf16* __restrict__ out) {
  const int row = blockIdx.x;
  const float* xr = x + (size_t)row * HDIM;
  float s = 0.f, ss = 0.f;
  for (int i = threadIdx.x; i < HDIM; i += 256) {
    float v = xr[i];
    s += v; ss += v * v;
  }
  __shared__ float r1[256], r2[256];
  r1[threadIdx.x] = s; r2[threadIdx.x] = ss;
  __syncthreads();
  for (int off = 128; off > 0; off >>= 1) {
    if (threadIdx.x < off) {
      r1[threadIdx.x] += r1[threadIdx.x + off];
      r2[threadIdx.x] += r2[threadIdx.x + off];
    }
    __syncthreads();
  }
  const float mu = r1[0] * (1.0f / HDIM);
  const float var = r2[0] * (1.0f / HDIM) - mu * mu;
  const float rstd = rsqrtf(var + 1e-5f);
  for (int i = threadIdx.x; i < HDIM; i += 256)
    out[(size_t)row * HDIM + i] = (bf16)((xr[i] - mu) * rstd * w[i] + b[i]);
}

__device__ __forceinline__ float gelu_tanh(float x) {
  const float c = 0.7978845608028654f;  // sqrt(2/pi)
  return 0.5f * x * (1.f + tanhf(c * (x + 0.044715f * x * x * x)));
}

__global__ void __launch_bounds__(256) router_logits_kernel(
    const bf16* __restrict__ ln2, const float* __restrict__ rw,
    float* __restrict__ logits) {
  int id = blockIdx.x * 256 + threadIdx.x;
  if (id >= TTOK * NEXP) return;
  const int t = id >> 3, e = id & 7;
  const bf16* xr = ln2 + (size_t)t * HDIM;
  const float* wr = rw + (size_t)e * HDIM;
  float acc = 0.f;
  for (int i = 0; i < HDIM; ++i) acc += (float)xr[i] * wr[i];
  logits[id] = acc;
}

__global__ void __launch_bounds__(256) gate_kernel(
    const float* __restrict__ logits, float* __restrict__ gate) {
  int t = blockIdx.x * 256 + threadIdx.x;
  if (t >= TTOK) return;
  float p[NEXP];
  float m = -3.402823466e38f;
  for (int e = 0; e < NEXP; ++e) { p[e] = logits[t * NEXP + e]; m = fmaxf(m, p[e]); }
  float sum = 0.f;
  for (int e = 0; e < NEXP; ++e) { p[e] = __expf(p[e] - m); sum += p[e]; }
  const float inv = 1.0f / sum;
  for (int e = 0; e < NEXP; ++e) p[e] *= inv;
  int i1 = 0;
  for (int e = 1; e < NEXP; ++e) if (p[e] > p[i1]) i1 = e;
  int i2 = (i1 == 0) ? 1 : 0;
  for (int e = 0; e < NEXP; ++e) if (e != i1 && p[e] > p[i2]) i2 = e;
  for (int e = 0; e < NEXP; ++e)
    gate[t * NEXP + e] = (e == i1) ? p[i1] : (e == i2) ? p[i2] : 0.f;
}

// ---------------- GEMM kernels ---------------------------------------------

// QKV: C[T,1536] = ln1 x qkvw^T; scatter q/k [b,h,s,d], V TRANSPOSED [b,h,d,s].
__global__ void __launch_bounds__(512) gemm_qkv_kernel(
    const bf16* __restrict__ A, const bf16* __restrict__ Bw,
    bf16* __restrict__ qb, bf16* __restrict__ kb, bf16* __restrict__ vtb) {
  extern __shared__ char smem[];
  const int rowBase = blockIdx.y * 128, colBase = blockIdx.x * 128;
  v8f acc[2][2] = {};
  gemm_core_128(A, HDIM, Bw, HDIM, HDIM, rowBase, colBase, lds_base(smem), acc);
  EPILOGUE_IDX
#pragma unroll
  for (int sm = 0; sm < 2; ++sm)
#pragma unroll
    for (int sn = 0; sn < 2; ++sn)
#pragma unroll
      for (int i = 0; i < 8; ++i) {
        const int t = rowBase + wm * 32 + sm * 16 + ml + i;
        const int n = colBase + wn * 32 + sn * 16 + nl;
        const int s = t / BATCH, b = t % BATCH;
        const bf16 val = (bf16)acc[sm][sn][i];
        if (n < NHEADS * HEADD) {
          const int h = n >> 6, d = n & 63;
          qb[(((size_t)b * NHEADS + h) * S_LEN + s) * HEADD + d] = val;
        } else if (n < (NHEADS + NKVH) * HEADD) {
          const int o = n - NHEADS * HEADD, h = o >> 6, d = o & 63;
          kb[(((size_t)b * NKVH + h) * S_LEN + s) * HEADD + d] = val;
        } else {
          const int o = n - (NHEADS + NKVH) * HEADD, h = o >> 6, d = o & 63;
          vtb[(((size_t)b * NKVH + h) * HEADD + d) * S_LEN + s] = val;  // V^T
        }
      }
}

// Proj: C = attn x projw^T; out = hidden + C (writes h_after and seeds d_out)
__global__ void __launch_bounds__(512) gemm_proj_kernel(
    const bf16* __restrict__ A, const bf16* __restrict__ Bw,
    const float* __restrict__ hidden, float* __restrict__ h_after,
    float* __restrict__ out) {
  extern __shared__ char smem[];
  const int rowBase = blockIdx.y * 128, colBase = blockIdx.x * 128;
  v8f acc[2][2] = {};
  gemm_core_128(A, HDIM, Bw, HDIM, HDIM, rowBase, colBase, lds_base(smem), acc);
  EPILOGUE_IDX
#pragma unroll
  for (int sm = 0; sm < 2; ++sm)
#pragma unroll
    for (int sn = 0; sn < 2; ++sn)
#pragma unroll
      for (int i = 0; i < 8; ++i) {
        const size_t idx =
            (size_t)(rowBase + wm * 32 + sm * 16 + ml + i) * HDIM +
            colBase + wn * 32 + sn * 16 + nl;
        const float v = hidden[idx] + acc[sm][sn][i];
        h_after[idx] = v;
        out[idx] = v;
      }
}

// FC1: a1[T,F] = gelu(ln2 x w1t^T), w1t = [F,H] bf16 (pre-transposed)
__global__ void __launch_bounds__(512) gemm_fc1_kernel(
    const bf16* __restrict__ A, const bf16* __restrict__ W1t,
    bf16* __restrict__ a1) {
  extern __shared__ char smem[];
  const int rowBase = blockIdx.y * 128, colBase = blockIdx.x * 128;
  v8f acc[2][2] = {};
  gemm_core_128(A, HDIM, W1t, HDIM, HDIM, rowBase, colBase, lds_base(smem), acc);
  EPILOGUE_IDX
#pragma unroll
  for (int sm = 0; sm < 2; ++sm)
#pragma unroll
    for (int sn = 0; sn < 2; ++sn)
#pragma unroll
      for (int i = 0; i < 8; ++i)
        a1[(size_t)(rowBase + wm * 32 + sm * 16 + ml + i) * FDIM +
           colBase + wn * 32 + sn * 16 + nl] = (bf16)gelu_tanh(acc[sm][sn][i]);
}

// FC2: out[T,H] += gate[t,e] * (a1 x w2t^T), w2t = [H,F] bf16 (pre-transposed)
__global__ void __launch_bounds__(512) gemm_fc2_kernel(
    const bf16* __restrict__ A, const bf16* __restrict__ W2t,
    const float* __restrict__ gate, int e, float* __restrict__ out) {
  extern __shared__ char smem[];
  const int rowBase = blockIdx.y * 128, colBase = blockIdx.x * 128;
  v8f acc[2][2] = {};
  gemm_core_128(A, FDIM, W2t, FDIM, FDIM, rowBase, colBase, lds_base(smem), acc);
  EPILOGUE_IDX
#pragma unroll
  for (int sm = 0; sm < 2; ++sm)
#pragma unroll
    for (int sn = 0; sn < 2; ++sn)
#pragma unroll
      for (int i = 0; i < 8; ++i) {
        const int t = rowBase + wm * 32 + sm * 16 + ml + i;
        const int n = colBase + wn * 32 + sn * 16 + nl;
        out[(size_t)t * HDIM + n] += gate[t * NEXP + e] * acc[sm][sn][i];
      }
}

// ---------------- Attention -------------------------------------------------
// Block = (b*NH+h, qtile); 512 thr / 16 waves. 64x1024 fp32 score strip lives
// in LDS (256KB of the WGP's 320KB). V is pre-transposed [b,kvh,d,s].
__global__ void __launch_bounds__(512) attn_kernel(
    const bf16* __restrict__ qb, const bf16* __restrict__ kb,
    const bf16* __restrict__ vtb, bf16* __restrict__ attn_out) {
  extern __shared__ char smem[];
  const uint32_t sbase = lds_base(smem);
  lds_float* sc = (lds_float*)(uintptr_t)sbase;          // [64][1024]
  const uint32_t qtOff = sbase + 64 * 1024 * 4;          // [64][64] bf16
  const uint32_t kvOff = qtOff + 64 * 64 * 2;            // [64][64] / [64][32]

  const int bh = blockIdx.x, qtile = blockIdx.y;
  const int b = bh / NHEADS, h = bh % NHEADS, kvh = h / (NHEADS / NKVH);
  const bf16* qg = qb + ((size_t)b * NHEADS + h) * S_LEN * HEADD;
  const bf16* kg = kb + ((size_t)b * NKVH + kvh) * S_LEN * HEADD;
  const bf16* vgt = vtb + ((size_t)b * NKVH + kvh) * HEADD * S_LEN;

  const int tid = threadIdx.x, lane = tid & 31, w = tid >> 5;
  const int wm = w >> 2, wn = w & 3;

  {  // async-stage Q tile (16B per thread)
    const int idx = tid * 8, r = idx >> 6, d = idx & 63;
    const bf16* gp = qg + (size_t)(qtile * 64 + r) * HEADD + d;
    const uint32_t la = qtOff + (uint32_t)(r * 128 + d * 2);
    asm volatile("global_load_async_to_lds_b128 %0, %1, off"
                 :: "v"(la), "v"(gp) : "memory");
    wait_async_lds();
  }

  // -------- pass 1: S = Q K^T * scale ------------------------------------
  for (int kt = 0; kt <= qtile; ++kt) {
    __syncthreads();
    {
      const int idx = tid * 8, r = idx >> 6, d = idx & 63;
      const bf16* gp = kg + (size_t)(kt * 64 + r) * HEADD + d;
      const uint32_t la = kvOff + (uint32_t)(r * 128 + d * 2);
      asm volatile("global_load_async_to_lds_b128 %0, %1, off"
                   :: "v"(la), "v"(gp) : "memory");
    }
    wait_async_lds();
    __syncthreads();
    v8f acc = {};
#pragma unroll
    for (int k0 = 0; k0 < 64; k0 += 32) {
      v16bf fa = frag_a(qtOff + (wm * 16 * 64 + k0) * 2, 64);
      v16bf fb = frag_b(kvOff + (wn * 16 * 64 + k0) * 2, 64);
      acc = wmma_bf16(fa, fb, acc);
    }
    const int n = wn * 16 + (lane & 15);
    const int mb = wm * 16 + ((lane & 16) ? 8 : 0);
#pragma unroll
    for (int i = 0; i < 8; ++i)
      sc[(mb + i) * 1024 + kt * 64 + n] = acc[i] * 0.125f;  // 1/sqrt(64)
  }
  __syncthreads();

  // -------- pass 2: causal softmax (8 threads / row) ----------------------
  {
    const int r = tid >> 3, sub = tid & 7;
    const int ncol = qtile * 64 + r + 1;
    float m = -3.402823466e38f;
    for (int c = sub; c < ncol; c += 8) m = fmaxf(m, sc[r * 1024 + c]);
#pragma unroll
    for (int off = 1; off < 8; off <<= 1) m = fmaxf(m, __shfl_xor(m, off, 32));
    float sum = 0.f;
    for (int c = sub; c < ncol; c += 8) {
      const float ev = __expf(sc[r * 1024 + c] - m);
      sc[r * 1024 + c] = ev;
      sum += ev;
    }
#pragma unroll
    for (int off = 1; off < 8; off <<= 1) sum += __shfl_xor(sum, off, 32);
    const float inv = 1.0f / sum;
    for (int c = sub; c < ncol; c += 8) sc[r * 1024 + c] *= inv;
    const int kend = (qtile + 1) * 64;
    for (int c = ncol + sub; c < kend; c += 8) sc[r * 1024 + c] = 0.f;
  }

  // -------- pass 3: O = P x V^T rows (K-contiguous) -----------------------
  v8f o = {};
  const int kend = (qtile + 1) * 64;
  for (int s0 = 0; s0 < kend; s0 += 32) {
    __syncthreads();
    {  // async-stage V^T[d][s0..s0+31]: 8B per thread
      const int idx = tid * 4, d = idx >> 5, si = idx & 31;
      const bf16* gp = vgt + (size_t)d * S_LEN + s0 + si;
      const uint32_t la = kvOff + (uint32_t)(d * 64 + si * 2);
      asm volatile("global_load_async_to_lds_b64 %0, %1, off"
                   :: "v"(la), "v"(gp) : "memory");
    }
    wait_async_lds();
    __syncthreads();
    const int m = lane & 15;
    const int koff = (lane & 16) ? 8 : 0;
    const lds_float* pr = sc + (wm * 16 + m) * 1024 + s0 + koff;
    v16bf fa;
#pragma unroll
    for (int e = 0; e < 8; ++e) fa[e] = (bf16)pr[e];
#pragma unroll
    for (int e = 0; e < 8; ++e) fa[8 + e] = (bf16)pr[16 + e];
    v16bf fb = frag_b(kvOff + (wn * 16 * 32) * 2, 32);
    o = wmma_bf16(fa, fb, o);
  }

  const int n = wn * 16 + (lane & 15);
  const int mb = wm * 16 + ((lane & 16) ? 8 : 0);
#pragma unroll
  for (int i = 0; i < 8; ++i) {
    const int s = qtile * 64 + mb + i;
    const int t = s * BATCH + b;
    attn_out[(size_t)t * HDIM + h * HEADD + n] = (bf16)o[i];
  }
}

// ---------------- Host-side launch ------------------------------------------

extern "C" void kernel_launch(void* const* d_in, const int* in_sizes, int n_in,
                              void* d_out, int out_size, void* d_ws,
                              size_t ws_size, hipStream_t stream) {
  (void)in_sizes; (void)n_in; (void)out_size; (void)ws_size;

  const float* hidden   = (const float*)d_in[0];
  const float* ln1_w    = (const float*)d_in[1];
  const float* ln1_b    = (const float*)d_in[2];
  const float* ln2_w    = (const float*)d_in[3];
  const float* ln2_b    = (const float*)d_in[4];
  const float* qkv_w    = (const float*)d_in[5];
  const float* proj_w   = (const float*)d_in[6];
  const float* router_w = (const float*)d_in[7];
  const float* moe_w1   = (const float*)d_in[8];
  const float* moe_w2   = (const float*)d_in[9];
  float* out = (float*)d_out;

  char* ws = (char*)d_ws;
  size_t off = 0;
  auto alloc = [&](size_t bytes) {
    void* p = ws + off;
    off = (off + bytes + 255) & ~(size_t)255;
    return p;
  };
  bf16* qkvw_bf = (bf16*)alloc((size_t)QKV_O * HDIM * 2);
  bf16* projw_bf = (bf16*)alloc((size_t)HDIM * HDIM * 2);
  bf16* w1t_bf = (bf16*)alloc((size_t)NEXP * HDIM * FDIM * 2);  // [E,F,H]
  bf16* w2t_bf = (bf16*)alloc((size_t)NEXP * FDIM * HDIM * 2);  // [E,H,F]
  bf16* ln1_bf = (bf16*)alloc((size_t)TTOK * HDIM * 2);
  bf16* q_bf = (bf16*)alloc((size_t)BATCH * NHEADS * S_LEN * HEADD * 2);
  bf16* k_bf = (bf16*)alloc((size_t)BATCH * NKVH * S_LEN * HEADD * 2);
  bf16* vt_bf = (bf16*)alloc((size_t)BATCH * NKVH * S_LEN * HEADD * 2);
  bf16* attn_bf = (bf16*)alloc((size_t)TTOK * HDIM * 2);
  float* h_after = (float*)alloc((size_t)TTOK * HDIM * 4);
  bf16* ln2_bf = (bf16*)alloc((size_t)TTOK * HDIM * 2);
  float* logits = (float*)alloc((size_t)TTOK * NEXP * 4);
  float* gate = (float*)alloc((size_t)TTOK * NEXP * 4);
  bf16* a1_bf = (bf16*)alloc((size_t)TTOK * FDIM * 2);

  auto cast = [&](const float* src, bf16* dst, int n) {
    cast_f32_bf16_kernel<<<(n + 255) / 256, 256, 0, stream>>>(src, dst, n);
  };
  cast(qkv_w, qkvw_bf, QKV_O * HDIM);
  cast(proj_w, projw_bf, HDIM * HDIM);
  // moe_w1 [E,H,F] -> w1t [E,F,H];  moe_w2 [E,F,H] -> w2t [E,H,F]
  transpose_cast_kernel<<<dim3(FDIM / 32, HDIM / 32, NEXP), 256, 0, stream>>>(
      moe_w1, w1t_bf, HDIM, FDIM);
  transpose_cast_kernel<<<dim3(HDIM / 32, FDIM / 32, NEXP), 256, 0, stream>>>(
      moe_w2, w2t_bf, FDIM, HDIM);

  layernorm_kernel<<<TTOK, 256, 0, stream>>>(hidden, ln1_w, ln1_b, ln1_bf);

  gemm_qkv_kernel<<<dim3(QKV_O / 128, TTOK / 128), 512, GEMM_LDS, stream>>>(
      ln1_bf, qkvw_bf, q_bf, k_bf, vt_bf);

  const size_t attn_lds = (size_t)64 * 1024 * 4 + 2 * (size_t)64 * 64 * 2;
  attn_kernel<<<dim3(BATCH * NHEADS, S_LEN / 64), 512, attn_lds, stream>>>(
      q_bf, k_bf, vt_bf, attn_bf);

  gemm_proj_kernel<<<dim3(HDIM / 128, TTOK / 128), 512, GEMM_LDS, stream>>>(
      attn_bf, projw_bf, hidden, h_after, out);

  layernorm_kernel<<<TTOK, 256, 0, stream>>>(h_after, ln2_w, ln2_b, ln2_bf);

  router_logits_kernel<<<(TTOK * NEXP + 255) / 256, 256, 0, stream>>>(
      ln2_bf, router_w, logits);
  gate_kernel<<<(TTOK + 255) / 256, 256, 0, stream>>>(logits, gate);

  for (int e = 0; e < NEXP; ++e) {
    gemm_fc1_kernel<<<dim3(FDIM / 128, TTOK / 128), 512, GEMM_LDS, stream>>>(
        ln2_bf, w1t_bf + (size_t)e * HDIM * FDIM, a1_bf);
    gemm_fc2_kernel<<<dim3(HDIM / 128, TTOK / 128), 512, GEMM_LDS, stream>>>(
        a1_bf, w2t_bf + (size_t)e * FDIM * HDIM, gate, e, out);
  }
}
